// S6_14010183320260
// MI455X (gfx1250) — matrix operand order repaired and checked by reading, not compiled, not run on hardware
//
#include <hip/hip_runtime.h>
#include <hip/hip_bf16.h>
#include <math.h>

// ---------------------------------------------------------------------------
// S6 selective-scan-like block, collapsed algebraically:
//   h0 == 0  =>  dA term vanishes (input A is dead)
//   y[b,l,d] = x[b,l,d] * softplus((X W1^T)[b,l,d] + b1[d]) * s[b,l]
//   s[b,l]   = sum_n (X W2^T + b2)[b,l,n] * (X W3^T + b3)[b,l,n]
// Dominant cost: 4096x1024x1024 GEMM -> done in BF16 WMMA (f32 accum),
// which puts it at the HBM roofline (~38MB traffic @ 23.3 TB/s).
// ---------------------------------------------------------------------------

typedef __attribute__((ext_vector_type(16))) __bf16 v16bf;
typedef __attribute__((ext_vector_type(8)))  float  v8f;

using u16 = unsigned short;
using u32 = unsigned int;

union FragBF { v16bf v; uint4 q[2]; };

static constexpr int D    = 1024;      // d_model (K and N of the big GEMM)
static constexpr int ROWS = 2 * 2048;  // B_SZ * SEQ_LEN = 4096 (M)

// ---------------------------------------------------------------------------
// f32 -> bf16 (round-to-nearest-even) conversion into workspace
// ---------------------------------------------------------------------------
__global__ void cvt_f32_to_bf16(const float* __restrict__ src,
                                u16* __restrict__ dst, int n) {
  int i = blockIdx.x * blockDim.x + threadIdx.x;
  if (i < n) {
    u32 u = __float_as_uint(src[i]);
    u32 r = (u + 0x7FFFu + ((u >> 16) & 1u)) >> 16;
    dst[i] = (u16)r;
  }
}

// ---------------------------------------------------------------------------
// Phase A: s[row] = dot(Bm[row,:], Cm[row,:]) over n=16, via WMMA.
// One wave handles 16 rows. A-frag: xb rows; B-frags: W2b/W3b (16 cols).
// ISA layouts (7.12.2):
//   A 16x32 bf16: lane<16 -> row=lane, K chunks {0..7, 16..23};
//                 lane>=16 -> row=lane-16, K chunks {8..15, 24..31}
//   B 32x16 bf16: lane holds one N column; this code loads A and B alike, K contiguous 16
//                 (lanes<16: K 0..15, lanes>=16: K 16..31)
//   C 16x16 f32:  VGPR r -> M=r (lanes 0-15) / M=r+8 (lanes 16-31), N=lane&15
// ---------------------------------------------------------------------------
__global__ void __launch_bounds__(256)
compute_s(const u16* __restrict__ xb, const u16* __restrict__ W2b,
          const u16* __restrict__ W3b, const float* __restrict__ b2,
          const float* __restrict__ b3, float* __restrict__ s) {
  const int lane = threadIdx.x & 31;
  const int wave = threadIdx.x >> 5;
  const int rowbase = (blockIdx.x * 8 + wave) * 16;
  const int l15 = lane & 15;
  const int ka = (lane & 16) ? 16 : 0;   // A k-chunk base within 32
  const int kb = (lane & 16) ? 16 : 0;   // B k-chunk base within 32

  const u16* Arow = xb  + (size_t)(rowbase + l15) * D;
  const u16* B2c  = W2b + (size_t)l15 * D;  // W2 row l15 = column l15 of B
  const u16* B3c  = W3b + (size_t)l15 * D;

  v8f accB = {};
  v8f accC = {};

  for (int k = 0; k < D; k += 32) {
    FragBF a, f2, f3;
    a.q[0]  = *reinterpret_cast<const uint4*>(Arow + k + ka);
    a.q[1]  = *reinterpret_cast<const uint4*>(Arow + k + ka + 8);
    f2.q[0] = *reinterpret_cast<const uint4*>(B2c + k + kb);
    f2.q[1] = *reinterpret_cast<const uint4*>(B2c + k + kb + 8);
    f3.q[0] = *reinterpret_cast<const uint4*>(B3c + k + kb);
    f3.q[1] = *reinterpret_cast<const uint4*>(B3c + k + kb + 8);
    accB = __builtin_amdgcn_wmma_f32_16x16x32_bf16(
        false, a.v, false, f2.v, (short)0, accB, false, false);
    accC = __builtin_amdgcn_wmma_f32_16x16x32_bf16(
        false, a.v, false, f3.v, (short)0, accC, false, false);
  }

  const float b2v = b2[l15];
  const float b3v = b3[l15];
#pragma unroll
  for (int r = 0; r < 8; ++r) {
    float p = (accB[r] + b2v) * (accC[r] + b3v);
    // reduce over the 16 N-lanes within each half-wave
    p += __shfl_xor(p, 1, 32);
    p += __shfl_xor(p, 2, 32);
    p += __shfl_xor(p, 4, 32);
    p += __shfl_xor(p, 8, 32);
    if (l15 == 0)
      s[rowbase + r + ((lane & 16) ? 8 : 0)] = p;
  }
}

// ---------------------------------------------------------------------------
// Phase B: y = x * softplus(X W1^T + b1) * s, fused into the big GEMM.
// Workgroup tile 128x128, 8 waves (4x2), each wave: 2x4 WMMA 16x16 tiles.
// ---------------------------------------------------------------------------
__global__ void __launch_bounds__(256)
mamba_fused(const float* __restrict__ x, const u16* __restrict__ xb,
            const u16* __restrict__ W1b, const float* __restrict__ b1,
            const float* __restrict__ s, float* __restrict__ y) {
  const int lane = threadIdx.x & 31;
  const int wave = threadIdx.x >> 5;
  const int mbase = blockIdx.x * 128 + (wave >> 1) * 32;  // 2 M-subtiles of 16
  const int nbase = blockIdx.y * 128 + (wave & 1) * 64;   // 4 N-subtiles of 16
  const int l15 = lane & 15;
  const int ka = (lane & 16) ? 16 : 0;
  const int kb = (lane & 16) ? 16 : 0;

  const u16* Arow0 = xb + (size_t)(mbase + l15) * D;
  const u16* Arow1 = xb + (size_t)(mbase + 16 + l15) * D;
  const u16* Bc0 = W1b + (size_t)(nbase + 0  + l15) * D;
  const u16* Bc1 = W1b + (size_t)(nbase + 16 + l15) * D;
  const u16* Bc2 = W1b + (size_t)(nbase + 32 + l15) * D;
  const u16* Bc3 = W1b + (size_t)(nbase + 48 + l15) * D;

  v8f acc[2][4] = {};

  for (int k = 0; k < D; k += 32) {
    FragBF a0, a1, b0, b1f, b2f, b3f;
    a0.q[0] = *reinterpret_cast<const uint4*>(Arow0 + k + ka);
    a0.q[1] = *reinterpret_cast<const uint4*>(Arow0 + k + ka + 8);
    a1.q[0] = *reinterpret_cast<const uint4*>(Arow1 + k + ka);
    a1.q[1] = *reinterpret_cast<const uint4*>(Arow1 + k + ka + 8);
    b0.q[0]  = *reinterpret_cast<const uint4*>(Bc0 + k + kb);
    b0.q[1]  = *reinterpret_cast<const uint4*>(Bc0 + k + kb + 8);
    b1f.q[0] = *reinterpret_cast<const uint4*>(Bc1 + k + kb);
    b1f.q[1] = *reinterpret_cast<const uint4*>(Bc1 + k + kb + 8);
    b2f.q[0] = *reinterpret_cast<const uint4*>(Bc2 + k + kb);
    b2f.q[1] = *reinterpret_cast<const uint4*>(Bc2 + k + kb + 8);
    b3f.q[0] = *reinterpret_cast<const uint4*>(Bc3 + k + kb);
    b3f.q[1] = *reinterpret_cast<const uint4*>(Bc3 + k + kb + 8);

    if (k + 32 < D) {  // stream the next K chunk (global_prefetch_b8)
      __builtin_prefetch(Arow0 + k + 32 + ka, 0, 1);
      __builtin_prefetch(Arow1 + k + 32 + ka, 0, 1);
      __builtin_prefetch(Bc0 + k + 32 + kb, 0, 1);
      __builtin_prefetch(Bc2 + k + 32 + kb, 0, 1);
    }

    acc[0][0] = __builtin_amdgcn_wmma_f32_16x16x32_bf16(
        false, a0.v, false, b0.v,  (short)0, acc[0][0], false, false);
    acc[0][1] = __builtin_amdgcn_wmma_f32_16x16x32_bf16(
        false, a0.v, false, b1f.v, (short)0, acc[0][1], false, false);
    acc[0][2] = __builtin_amdgcn_wmma_f32_16x16x32_bf16(
        false, a0.v, false, b2f.v, (short)0, acc[0][2], false, false);
    acc[0][3] = __builtin_amdgcn_wmma_f32_16x16x32_bf16(
        false, a0.v, false, b3f.v, (short)0, acc[0][3], false, false);
    acc[1][0] = __builtin_amdgcn_wmma_f32_16x16x32_bf16(
        false, a1.v, false, b0.v,  (short)0, acc[1][0], false, false);
    acc[1][1] = __builtin_amdgcn_wmma_f32_16x16x32_bf16(
        false, a1.v, false, b1f.v, (short)0, acc[1][1], false, false);
    acc[1][2] = __builtin_amdgcn_wmma_f32_16x16x32_bf16(
        false, a1.v, false, b2f.v, (short)0, acc[1][2], false, false);
    acc[1][3] = __builtin_amdgcn_wmma_f32_16x16x32_bf16(
        false, a1.v, false, b3f.v, (short)0, acc[1][3], false, false);
  }

  // Epilogue: y[row, col] = x[row, col] * softplus(acc + b1[col]) * s[row]
  float b1v[4];
#pragma unroll
  for (int ni = 0; ni < 4; ++ni) b1v[ni] = b1[nbase + ni * 16 + l15];

#pragma unroll
  for (int mi = 0; mi < 2; ++mi) {
#pragma unroll
    for (int ni = 0; ni < 4; ++ni) {
      const int col = nbase + ni * 16 + l15;
#pragma unroll
      for (int r = 0; r < 8; ++r) {
        const int row = mbase + mi * 16 + r + ((lane & 16) ? 8 : 0);
        float z = acc[mi][ni][r] + b1v[ni];
        float sp = (z > 20.0f) ? z : log1pf(__expf(z));  // softplus
        const size_t idx = (size_t)row * D + col;
        y[idx] = x[idx] * sp * s[row];
      }
    }
  }
}

// ---------------------------------------------------------------------------
// Launch
// ---------------------------------------------------------------------------
extern "C" void kernel_launch(void* const* d_in, const int* in_sizes, int n_in,
                              void* d_out, int out_size, void* d_ws, size_t ws_size,
                              hipStream_t stream) {
  (void)in_sizes; (void)n_in; (void)out_size; (void)ws_size;

  const float* x  = (const float*)d_in[0];  // [2,2048,1024]
  const float* W1 = (const float*)d_in[1];  // [1024,1024]
  const float* b1 = (const float*)d_in[2];  // [1024]
  const float* W2 = (const float*)d_in[3];  // [16,1024]
  const float* b2 = (const float*)d_in[4];  // [16]
  const float* W3 = (const float*)d_in[5];  // [16,1024]
  const float* b3 = (const float*)d_in[6];  // [16]
  // d_in[7] = A: unused (h0 == 0 kills the dA*h term exactly)
  float* y = (float*)d_out;

  // workspace layout (bytes)
  char* ws = (char*)d_ws;
  u16* xb  = (u16*)(ws);                                   // 4194304 bf16
  u16* W1b = (u16*)(ws + 8388608);                         // 1048576 bf16
  u16* W2b = (u16*)(ws + 8388608 + 2097152);               //   16384 bf16
  u16* W3b = (u16*)(ws + 8388608 + 2097152 + 32768);       //   16384 bf16
  float* s = (float*)(ws + 8388608 + 2097152 + 65536);     //    4096 f32

  const int nx = ROWS * D;        // 4194304
  const int nw1 = D * D;          // 1048576
  const int nw23 = 16 * D;        // 16384

  cvt_f32_to_bf16<<<(nx + 255) / 256, 256, 0, stream>>>(x, xb, nx);
  cvt_f32_to_bf16<<<(nw1 + 255) / 256, 256, 0, stream>>>(W1, W1b, nw1);
  cvt_f32_to_bf16<<<(nw23 + 255) / 256, 256, 0, stream>>>(W2, W2b, nw23);
  cvt_f32_to_bf16<<<(nw23 + 255) / 256, 256, 0, stream>>>(W3, W3b, nw23);

  compute_s<<<ROWS / (8 * 16), 256, 0, stream>>>(xb, W2b, W3b, b2, b3, s);

  dim3 grid(ROWS / 128, D / 128);  // (32, 8)
  mamba_fused<<<grid, 256, 0, stream>>>(x, xb, W1b, b1, s, y);
}
